// Head_9088150798882
// MI455X (gfx1250) — matrix operand order repaired and checked by reading, not compiled, not run on hardware
//
#include <hip/hip_runtime.h>

typedef __attribute__((ext_vector_type(16))) _Float16 v16h;
typedef __attribute__((ext_vector_type(8)))  _Float16 v8h;
typedef __attribute__((ext_vector_type(8)))  float    v8f;
typedef __attribute__((ext_vector_type(4)))  unsigned u32x4;
typedef __attribute__((ext_vector_type(8)))  int      i32x8;
typedef __attribute__((ext_vector_type(4)))  int      i32x4;

#define SCALE_QK 0.05103103630798287f  /* 384^-0.5 (reference scales by n_embd) */

__device__ __forceinline__ v16h comb(v8h lo, v8h hi) {
  v16h r;
#pragma unroll
  for (int e = 0; e < 8; ++e) { r[e] = lo[e]; r[e + 8] = hi[e]; }
  return r;
}

__device__ __forceinline__ v8f wmma_f16(v16h a, v16h b, v8f c) {
  // D = A(16x32 f16) * B(32x16 f16) + C(16x16 f32)
  return __builtin_amdgcn_wmma_f32_16x16x32_f16(false, a, false, b, (short)0, c,
                                                false, false);
}

// ---------------------------------------------------------------------------
// Tensor Data Mover: 2D tile load global -> LDS with hardware row padding.
// Descriptor bitfields per CDNA5 ISA §8.3 (group0) / §8.4 (group1).
//   dims/strides in elements (data_size = 2 bytes, f16).
//   padc: pad_interval code (4 -> every 32 dwords, 6 -> every 128 dwords)
//   pada: pad_amount encoded (3 -> 4 dwords = 8 f16 of padding)
// ---------------------------------------------------------------------------
__device__ __forceinline__ void tdm_load_2d(unsigned lds_off, const void* gptr,
                                            unsigned dim0, unsigned dim1,
                                            unsigned stride0,
                                            unsigned padc, unsigned pada)
{
  const unsigned long long ga = (unsigned long long)(size_t)gptr;
  u32x4 g0;
  g0[0] = 1u;                                   // count=1, no gather
  g0[1] = lds_off;                              // lds_addr        [63:32]
  g0[2] = (unsigned)(ga & 0xffffffffu);         // global_addr lo  [95:64]
  g0[3] = (unsigned)((ga >> 32) & 0x1ffffffu)   // global_addr hi  [120:96]
        | (2u << 30);                           // type=2 ("image")[127:126]
  i32x8 g1;
  g1[0] = (int)((1u << 16)                      // data_size = 2B  [17:16]
              | (1u << 20)                      // pad_enable      [20]
              | (padc << 22)                    // pad_interval    [24:22]
              | (pada << 25));                  // pad_amount      [31:25]
  g1[1] = (int)(dim0 << 16);                    // tensor_dim0     [79:48]
  g1[2] = (int)((dim0 >> 16) | (dim1 << 16));   // tensor_dim1     [111:80]
  g1[3] = (int)((dim1 >> 16) | (dim0 << 16));   // tile_dim0=dim0  [127:112]
  g1[4] = (int)(dim1 & 0xffffu);                // tile_dim1=dim1, tile_dim2=0
  g1[5] = (int)stride0;                         // tensor_dim0_stride lo
  g1[6] = 0;                                    // stride0 hi | stride1 lo
  g1[7] = 0;
  const i32x4 gz = {};
#if defined(__clang_major__) && __clang_major__ >= 23
  const i32x8 gz8 = {};
  __builtin_amdgcn_tensor_load_to_lds(g0, g1, gz, gz, gz8, 0);
#else
  __builtin_amdgcn_tensor_load_to_lds(g0, g1, gz, gz, 0);
#endif
}

// ---------------------------------------------------------------------------
// Kernel 1: projection GEMM.  x:(BT=131072, 384) f32  @  W:(384,64) f32
//   -> q/k:(b,t,h) f16, v:(b,h,t) f16 (transposed for TDM staging).
//   grid = (1024, 3); block = 256 (8 waves); 128 rows x 64 cols per block.
// ---------------------------------------------------------------------------
__global__ __launch_bounds__(256) void proj_kernel(
    const float* __restrict__ x,
    const float* __restrict__ Wq, const float* __restrict__ Wk,
    const float* __restrict__ Wv,
    _Float16* __restrict__ qo, _Float16* __restrict__ ko,
    _Float16* __restrict__ vo)
{
  constexpr int C = 384, H = 64, XS = C + 8;       // padded f16 stride
  __shared__ _Float16 Xs[128 * XS];                // 100,352 B
  __shared__ _Float16 Wt[H * XS];                  //  50,176 B (W transposed)

  const int tid = threadIdx.x;
  const int wy  = blockIdx.y;
  const float*   W    = (wy == 0) ? Wq : (wy == 1) ? Wk : Wv;
  _Float16*      outp = (wy == 0) ? qo : ko;
  const float    osc  = (wy == 0) ? SCALE_QK : 1.0f;
  const long rowBase = (long)blockIdx.x * 128;

  // Stage X tile (f32 -> f16), 128x384, float4 reads, 8B LDS writes.
  for (int i = tid; i < 128 * (C / 4); i += 256) {
    const int r  = i / (C / 4);
    const int c4 = (i % (C / 4)) * 4;
    const float4 f = *(const float4*)(x + (rowBase + r) * C + c4);
    union { _Float16 h[4]; uint2 u; } p;
    p.h[0] = (_Float16)f.x; p.h[1] = (_Float16)f.y;
    p.h[2] = (_Float16)f.z; p.h[3] = (_Float16)f.w;
    *(uint2*)&Xs[r * XS + c4] = p.u;
  }
  // Stage W transposed: Wt[n][k] = W[k][n]  (coalesced global reads over n).
  for (int i = tid; i < H * C; i += 256) {
    const int kk = i / H, n = i % H;
    Wt[n * XS + kk] = (_Float16)W[(long)kk * H + n];
  }
  __syncthreads();

  const int lane = tid & 31, wave = tid >> 5;
  const int ln = lane & 15, half = lane >> 4;

  // A fragments: 12 K-chunks of 32 for this wave's 16 rows.
  v16h afr[12];
#pragma unroll
  for (int kc = 0; kc < 12; ++kc) {
    const _Float16* pa = &Xs[(wave * 16 + ln) * XS + kc * 32 + half * 8];
    afr[kc] = comb(*(const v8h*)pa, *(const v8h*)(pa + 16));
  }
#pragma unroll
  for (int nt = 0; nt < 4; ++nt) {
    // Preload all 12 B fragments, then issue the 12 WMMAs back-to-back.
    const _Float16* bb = &Wt[(nt * 16 + ln) * XS + half * 8];
    v16h bfr[12];
#pragma unroll
    for (int kc = 0; kc < 12; ++kc)
      bfr[kc] = comb(*(const v8h*)(bb + kc * 32), *(const v8h*)(bb + kc * 32 + 16));
    v8f acc = {};
#pragma unroll
    for (int kc = 0; kc < 12; ++kc)
      acc = wmma_f16(afr[kc], bfr[kc], acc);

    const int h = nt * 16 + ln;
    if (wy == 2) {                       // V: store transposed (b, h, t)
      const long b = rowBase >> 8;
      const int t0 = (int)(rowBase & 255) + wave * 16;
#pragma unroll
      for (int r = 0; r < 8; ++r)
        vo[(b * 64 + h) * 256 + t0 + r + half * 8] = (_Float16)acc[r];
    } else {                             // Q (pre-scaled) / K: (b, t, h)
#pragma unroll
      for (int r = 0; r < 8; ++r) {
        const long row = rowBase + wave * 16 + r + half * 8;
        outp[row * H + h] = (_Float16)(acc[r] * osc);
      }
    }
  }
}

// ---------------------------------------------------------------------------
// Kernel 2: fused causal flash attention per batch.  grid = 512, block = 256.
//   Staging via three TDM descriptor loads (hardware row padding), then each
//   wave handles 32 query rows (two 16-row M-tiles) with online softmax.
// ---------------------------------------------------------------------------
__global__ __launch_bounds__(256) void attn_kernel(
    const _Float16* __restrict__ q, const _Float16* __restrict__ k,
    const _Float16* __restrict__ v, float* __restrict__ out)
{
  constexpr int T = 256, H = 64, KS = H + 8, VS = T + 8, PS = 40;
  __shared__ _Float16 Qs[T * KS];        // 36,864 B  (row stride 72 f16)
  __shared__ _Float16 Ks[T * KS];        // 36,864 B
  __shared__ _Float16 Vt[H * VS];        // 33,792 B  (V^T: [h][t], stride 264)
  __shared__ _Float16 Ps[8 * 16 * PS];   // 10,240 B  (per-wave P stage)

  const int tid = threadIdx.x;
  const long base = (long)blockIdx.x * T * H;

  if (tid < 32) {  // wave 0 issues the DMA; TDM ignores EXEC, one per wave
    // Q/K: 64-elem rows x 256; pad 4 dwords after every 32 dwords -> stride 72h
    tdm_load_2d((unsigned)(size_t)&Qs[0], q + base, H, T, H, 4u, 3u);
    tdm_load_2d((unsigned)(size_t)&Ks[0], k + base, H, T, H, 4u, 3u);
    // V^T: 256-elem rows x 64; pad 4 dwords after every 128 dwords -> 264h
    tdm_load_2d((unsigned)(size_t)&Vt[0], v + base, T, H, T, 6u, 3u);
    __builtin_amdgcn_s_wait_tensorcnt(0);
  }
  __syncthreads();

  const int lane = tid & 31, wave = tid >> 5;
  const int ln = lane & 15, half = lane >> 4;
  _Float16* Pw = &Ps[wave * 16 * PS];

  for (int mt = 0; mt < 2; ++mt) {
    const int MT = wave * 2 + mt;        // 16-row M-tile index, 0..15
    const int qrow = MT * 16;

    v16h qa[2];                          // Q A-fragments over H=64
#pragma unroll
    for (int kc = 0; kc < 2; ++kc) {
      const _Float16* pa = &Qs[(qrow + ln) * KS + kc * 32 + half * 8];
      qa[kc] = comb(*(const v8h*)pa, *(const v8h*)(pa + 16));
    }

    v8f o0 = {}, o1 = {}, o2 = {}, o3 = {};
    float rmax[8], rsum[8];
#pragma unroll
    for (int r = 0; r < 8; ++r) { rmax[r] = -3.0e38f; rsum[r] = 0.0f; }

    const int ncMax = MT >> 1;           // key chunks of 32 up to diagonal
    for (int nc = 0; nc <= ncMax; ++nc) {
      // ---- S = Q @ K^T: preload 4 B-fragments, then 4 WMMAs ----
      const _Float16* bb0 = &Ks[(nc * 32 +  0 + ln) * KS + half * 8];
      const _Float16* bb1 = &Ks[(nc * 32 + 16 + ln) * KS + half * 8];
      v16h bk[4];
      bk[0] = comb(*(const v8h*)(bb0),      *(const v8h*)(bb0 + 16));
      bk[1] = comb(*(const v8h*)(bb0 + 32), *(const v8h*)(bb0 + 48));
      bk[2] = comb(*(const v8h*)(bb1),      *(const v8h*)(bb1 + 16));
      bk[3] = comb(*(const v8h*)(bb1 + 32), *(const v8h*)(bb1 + 48));
      v8f s0 = {}, s1 = {};
      s0 = wmma_f16(qa[0], bk[0], s0);
      s0 = wmma_f16(qa[1], bk[1], s0);
      s1 = wmma_f16(qa[0], bk[2], s1);
      s1 = wmma_f16(qa[1], bk[3], s1);
      // ---- causal mask (q was pre-scaled by 384^-0.5) ----
#pragma unroll
      for (int r = 0; r < 8; ++r) {
        const int qi  = qrow + r + half * 8;
        const int kj0 = nc * 32 + ln;
        if (kj0      > qi) s0[r] = -3.0e38f;
        if (kj0 + 16 > qi) s1[r] = -3.0e38f;
      }
      // ---- online softmax: rows live in one 16-lane half per VGPR ----
      float alpha[8];
#pragma unroll
      for (int r = 0; r < 8; ++r) {
        float m = fmaxf(s0[r], s1[r]);
#pragma unroll
        for (int d = 1; d < 16; d <<= 1) m = fmaxf(m, __shfl_xor(m, d, 32));
        const float nm = fmaxf(rmax[r], m);
        const float a  = __expf(rmax[r] - nm);
        rmax[r] = nm; alpha[r] = a;
        const float p0 = __expf(s0[r] - nm);
        const float p1 = __expf(s1[r] - nm);
        s0[r] = p0; s1[r] = p1;
        float ps = p0 + p1;
#pragma unroll
        for (int d = 1; d < 16; d <<= 1) ps += __shfl_xor(ps, d, 32);
        rsum[r] = rsum[r] * a + ps;
      }
#pragma unroll
      for (int r = 0; r < 8; ++r) {
        o0[r] *= alpha[r]; o1[r] *= alpha[r];
        o2[r] *= alpha[r]; o3[r] *= alpha[r];
      }
      // ---- stage P (C-layout -> row-major LDS -> A-fragment) ----
#pragma unroll
      for (int r = 0; r < 8; ++r) {
        const int prow = r + half * 8;
        Pw[prow * PS + ln]      = (_Float16)s0[r];
        Pw[prow * PS + 16 + ln] = (_Float16)s1[r];
      }
      asm volatile("s_wait_dscnt 0" ::: "memory");
      const _Float16* pp = &Pw[ln * PS + half * 8];
      const v16h pfr = comb(*(const v8h*)pp, *(const v8h*)(pp + 16));
      // ---- O += P @ V: preload 4 V-fragments, then 4 WMMAs ----
      v16h vf[4];
#pragma unroll
      for (int ht = 0; ht < 4; ++ht) {
        const _Float16* vb = &Vt[(ht * 16 + ln) * VS + nc * 32 + half * 8];
        vf[ht] = comb(*(const v8h*)vb, *(const v8h*)(vb + 16));
      }
      o0 = wmma_f16(pfr, vf[0], o0);
      o1 = wmma_f16(pfr, vf[1], o1);
      o2 = wmma_f16(pfr, vf[2], o2);
      o3 = wmma_f16(pfr, vf[3], o3);
    }
    // ---- finalize: divide by row sums, store f32 ----
#pragma unroll
    for (int r = 0; r < 8; ++r) {
      const float inv = 1.0f / rsum[r];
      const int row = qrow + r + half * 8;
      float* op = out + base + (long)row * H;
      op[ 0 + ln] = o0[r] * inv;
      op[16 + ln] = o1[r] * inv;
      op[32 + ln] = o2[r] * inv;
      op[48 + ln] = o3[r] * inv;
    }
  }
}

extern "C" void kernel_launch(void* const* d_in, const int* in_sizes, int n_in,
                              void* d_out, int out_size, void* d_ws, size_t ws_size,
                              hipStream_t stream)
{
  (void)in_sizes; (void)n_in; (void)out_size; (void)ws_size;
  // setup_inputs order: x, Wk, Wq, Wv
  const float* x  = (const float*)d_in[0];
  const float* Wk = (const float*)d_in[1];
  const float* Wq = (const float*)d_in[2];
  const float* Wv = (const float*)d_in[3];
  float* out = (float*)d_out;

  const size_t BT_H = (size_t)512 * 256 * 64;     // 8,388,608 elements
  _Float16* qb = (_Float16*)d_ws;                 // 16 MB each, 48 MB total
  _Float16* kb = qb + BT_H;
  _Float16* vb = kb + BT_H;                       // v stored (b, h, t)

  proj_kernel<<<dim3(1024, 3, 1), dim3(256, 1, 1), 0, stream>>>(
      x, Wq, Wk, Wv, qb, kb, vb);
  attn_kernel<<<dim3(512, 1, 1), dim3(256, 1, 1), 0, stream>>>(
      qb, kb, vb, out);
}
